// GroupedQueryAttention_58823872086521
// MI455X (gfx1250) — matrix operand order repaired and checked by reading, not compiled, not run on hardware
//
#include <hip/hip_runtime.h>
#include <hip/hip_bf16.h>

typedef __attribute__((ext_vector_type(16))) _Float16 v16h;
typedef __attribute__((ext_vector_type(8)))  _Float16 v8h;
typedef __attribute__((ext_vector_type(8)))  float    v8f;

union F16Frag { v16h v; v8h h[2]; };

#define B_   2
#define T_   2048
#define DM_  2048
#define NH_  16
#define NKV_ 4
#define DH_  128
#define NV_  256
#define NEGBIG (-1000000000.0f)
#define SCALE_ (0.08838834764831845f)   // 1/sqrt(128)

static __device__ __forceinline__ v8f wmma_f16(v16h a, v16h b, v8f c) {
  // D(16x16 f32) = A(16x32 f16) * B(32x16 f16) + C
  return __builtin_amdgcn_wmma_f32_16x16x32_f16(false, a, false, b, (short)0, c, false, false);
}

// Async global->LDS copy of 16 bytes (CDNA5, ASYNCcnt-tracked).
static __device__ __forceinline__ void async_copy_b128(unsigned lds_byte_addr,
                                                       const void* gaddr) {
  unsigned long long ga = (unsigned long long)(uintptr_t)gaddr;
  asm volatile("global_load_async_to_lds_b128 %0, %1, off"
               :: "v"(lds_byte_addr), "v"(ga) : "memory");
}
static __device__ __forceinline__ void async_wait0() {
  asm volatile("s_wait_asynccnt 0x0" ::: "memory");
}

// ---------------------------------------------------------------- casts
__global__ void cast_f32_to_f16(const float* __restrict__ in, _Float16* __restrict__ out, int n) {
  int idx = blockIdx.x * blockDim.x + threadIdx.x;
  if (idx < n) out[idx] = (_Float16)in[idx];
}

// in: (K,N) f32 row-major  ->  out: (N,K) f16 row-major (transposed)
__global__ void castT_f32_to_f16(const float* __restrict__ in, _Float16* __restrict__ out, int K, int N) {
  int idx = blockIdx.x * blockDim.x + threadIdx.x;
  if (idx >= K * N) return;
  int k = idx / N, n = idx - k * N;
  out[(size_t)n * K + k] = (_Float16)in[idx];
}

// ---------------------------------------------------------------- register-blocked WMMA GEMM
// C(M,N) f32 = A(M,K) f16 * Bt(N,K)^T f16.
// 256 threads = 8 waves; wave tile 32x64 (2 A-frags x 4 B-frags -> 8 WMMA / k-step);
// block tile 64x256. Requires M%64==0, N%256==0, K%32==0.
__global__ __launch_bounds__(256) void gemm_f16_wmma(const _Float16* __restrict__ A,
                                                     const _Float16* __restrict__ Bt,
                                                     float* __restrict__ C,
                                                     int M, int N, int K) {
  const int wave = threadIdx.x >> 5;
  const int lane = threadIdx.x & 31;
  const int wr = wave >> 2, wc = wave & 3;
  const int tm = blockIdx.y * 64 + wr * 32;
  const int tn = blockIdx.x * 256 + wc * 64;
  const int lm = lane & 15;
  const int hi = lane >> 4;
  const int kb = hi * 16;

  const _Float16* ar0 = A + (size_t)(tm + lm) * K;
  const _Float16* ar1 = A + (size_t)(tm + 16 + lm) * K;
  const _Float16* br0 = Bt + (size_t)(tn + lm) * K;
  const _Float16* br1 = Bt + (size_t)(tn + 16 + lm) * K;
  const _Float16* br2 = Bt + (size_t)(tn + 32 + lm) * K;
  const _Float16* br3 = Bt + (size_t)(tn + 48 + lm) * K;

  v8f acc[2][4];
  #pragma unroll
  for (int i = 0; i < 2; i++)
    #pragma unroll
    for (int j = 0; j < 4; j++) acc[i][j] = (v8f){};

  for (int k0 = 0; k0 < K; k0 += 32) {
    F16Frag a0, a1;
    {
      const v8h* p0 = (const v8h*)(ar0 + k0 + kb);
      a0.h[0] = p0[0]; a0.h[1] = p0[1];
      const v8h* p1 = (const v8h*)(ar1 + k0 + kb);
      a1.h[0] = p1[0]; a1.h[1] = p1[1];
    }
    v16h b0 = *(const v16h*)(br0 + k0 + hi * 16);
    v16h b1 = *(const v16h*)(br1 + k0 + hi * 16);
    v16h b2 = *(const v16h*)(br2 + k0 + hi * 16);
    v16h b3 = *(const v16h*)(br3 + k0 + hi * 16);

    acc[0][0] = wmma_f16(a0.v, b0, acc[0][0]);
    acc[0][1] = wmma_f16(a0.v, b1, acc[0][1]);
    acc[0][2] = wmma_f16(a0.v, b2, acc[0][2]);
    acc[0][3] = wmma_f16(a0.v, b3, acc[0][3]);
    acc[1][0] = wmma_f16(a1.v, b0, acc[1][0]);
    acc[1][1] = wmma_f16(a1.v, b1, acc[1][1]);
    acc[1][2] = wmma_f16(a1.v, b2, acc[1][2]);
    acc[1][3] = wmma_f16(a1.v, b3, acc[1][3]);
  }

  #pragma unroll
  for (int i = 0; i < 2; i++) {
    float* crow = C + (size_t)(tm + i * 16 + hi * 8) * N + tn + lm;
    #pragma unroll
    for (int j = 0; j < 4; j++)
      #pragma unroll
      for (int r = 0; r < 8; r++)
        crow[(size_t)r * N + j * 16] = acc[i][j][r];
  }
}

// ---------------------------------------------------------------- RoPE (Q) + f16 pack
__global__ void rope_q_kernel(const float* __restrict__ q, const float* __restrict__ cosb,
                              const float* __restrict__ sinb, _Float16* __restrict__ qh) {
  int idx = blockIdx.x * blockDim.x + threadIdx.x;   // B*T*NH*64
  int i = idx & 63;
  int h = (idx >> 6) & (NH_ - 1);
  int t = (idx >> 10) & (T_ - 1);
  int b = idx >> 21;
  if (b >= B_) return;
  size_t base = ((size_t)(b * T_ + t)) * DM_ + h * DH_ + 2 * i;
  float a = q[base], bb = q[base + 1];
  float ra = a, rb = bb;
  if (t >= NV_) {
    float c = cosb[(size_t)(t - NV_) * 64 + i];
    float s = sinb[(size_t)(t - NV_) * 64 + i];
    ra = a * c - bb * s;
    rb = a * s + bb * c;
  }
  qh[base]     = (_Float16)ra;
  qh[base + 1] = (_Float16)rb;
}

// ---------------------------------------------------------------- RoPE (K) + f16 pack + f32 cache
__global__ void rope_k_kernel(const float* __restrict__ k, const float* __restrict__ cosb,
                              const float* __restrict__ sinb, _Float16* __restrict__ kh,
                              float* __restrict__ kcache) {
  int idx = blockIdx.x * blockDim.x + threadIdx.x;   // B*T*NKV*64
  int i = idx & 63;
  int h = (idx >> 6) & (NKV_ - 1);
  int t = (idx >> 8) & (T_ - 1);
  int b = idx >> 19;
  if (b >= B_) return;
  size_t base = ((size_t)(b * T_ + t)) * (NKV_ * DH_) + h * DH_ + 2 * i;
  float a = k[base], bb = k[base + 1];
  float ra = a, rb = bb;
  if (t >= NV_) {
    float c = cosb[(size_t)(t - NV_) * 64 + i];
    float s = sinb[(size_t)(t - NV_) * 64 + i];
    ra = a * c - bb * s;
    rb = a * s + bb * c;
  }
  kh[base]     = (_Float16)ra;
  kh[base + 1] = (_Float16)rb;
  size_t cb = (((size_t)b * NKV_ + h) * T_ + t) * DH_ + 2 * i;   // (B,NKV,T,DH) cache
  kcache[cb]     = ra;
  kcache[cb + 1] = rb;
}

// ---------------------------------------------------------------- V: transpose-pack f16 + f32 cache
__global__ void v_pack_kernel(const float* __restrict__ v, _Float16* __restrict__ vt,
                              float* __restrict__ vcache) {
  int idx = blockIdx.x * blockDim.x + threadIdx.x;   // B*T*NKV*DH
  int d = idx & (DH_ - 1);
  int h = (idx >> 7) & (NKV_ - 1);
  int t = (idx >> 9) & (T_ - 1);
  int b = idx >> 20;
  if (b >= B_) return;
  float val = v[((size_t)(b * T_ + t)) * (NKV_ * DH_) + h * DH_ + d];
  vt[(((size_t)b * NKV_ + h) * DH_ + d) * T_ + t] = (_Float16)val;   // [b][hk][d][t], t-contig
  vcache[(((size_t)b * NKV_ + h) * T_ + t) * DH_ + d] = val;         // (B,NKV,T,DH) cache
}

// ---------------------------------------------------------------- flash attention
// grid: (T/128, NH, B); block 256 (8 waves). Wave w owns a 16x128 Q tile.
// K/V key-tiles (32 keys) staged in LDS once per block via async global->LDS copies.
__global__ __launch_bounds__(256) void attn_kernel(const _Float16* __restrict__ qh,
                                                   const _Float16* __restrict__ kh,
                                                   const _Float16* __restrict__ vt,
                                                   _Float16* __restrict__ attn_h) {
  __shared__ __align__(16) _Float16 sK[32][128];    // [key][d]   8 KB
  __shared__ __align__(16) _Float16 sVt[128][32];   // [d][key]   8 KB
  __shared__ __align__(16) _Float16 sP[8][16 * 32]; // per-wave P 8 KB

  const int qchunk = blockIdx.x;
  const int h = blockIdx.y;
  const int b = blockIdx.z;
  const int wave = threadIdx.x >> 5;
  const int lane = threadIdx.x & 31;
  const int qb = qchunk * 128 + wave * 16;
  const int hk = h >> 2;                 // N_REP = 4
  const int lm = lane & 15;
  const int hi = lane >> 4;
  const int kb = hi * 16;

  const unsigned sK_base  = (unsigned)(uintptr_t)(&sK[0][0]);
  const unsigned sVt_base = (unsigned)(uintptr_t)(&sVt[0][0]);

  // Preload Q A-fragments: 16 rows x 128 d -> 4 frags of 16x32
  F16Frag qf[4];
  const _Float16* qrow = qh + ((size_t)(b * T_) + qb + lm) * DM_ + h * DH_;
  #pragma unroll
  for (int f = 0; f < 4; f++) {
    const v8h* p = (const v8h*)(qrow + f * 32 + kb);
    qf[f].h[0] = p[0];
    qf[f].h[1] = p[1];
  }

  v8f o[8];
  #pragma unroll
  for (int f = 0; f < 8; f++) o[f] = (v8f){};
  float mrow[8], lrow[8];
  #pragma unroll
  for (int r = 0; r < 8; r++) { mrow[r] = -1e30f; lrow[r] = 0.0f; }

  // Vision chunks (rows < NV) attend to all keys; text chunks stop at diagonal.
  const int ktend = (qchunk * 128 < NV_) ? (T_ / 32) : ((qchunk * 128 + 128) / 32);

  const _Float16* kbase_p = kh + (size_t)(b * T_) * (NKV_ * DH_) + hk * DH_;
  const _Float16* vbase_p = vt + ((size_t)(b * NKV_) + hk) * DH_ * (size_t)T_;

  for (int kt = 0; kt < ktend; kt++) {
    // ---- stage K tile (32x128) and V^T tile (128x32) into LDS, async
    {
      const _Float16* kg = kbase_p + (size_t)(kt * 32) * (NKV_ * DH_);
      #pragma unroll
      for (int it = 0; it < 2; it++) {
        int t = threadIdx.x + it * 256;            // 512 b128 transfers for K
        int row = t >> 4, seg = t & 15;
        async_copy_b128(sK_base + row * 256 + seg * 16,
                        (const char*)(kg + (size_t)row * (NKV_ * DH_)) + seg * 16);
      }
      const _Float16* vg = vbase_p + kt * 32;
      #pragma unroll
      for (int it = 0; it < 2; it++) {
        int t = threadIdx.x + it * 256;            // 512 b128 transfers for V^T
        int d = t >> 2, seg = t & 3;
        async_copy_b128(sVt_base + d * 64 + seg * 16,
                        (const char*)(vg + (size_t)d * T_) + seg * 16);
      }
      async_wait0();
      __syncthreads();
    }

    // ---- S = Q * K^T (two 16x16 tiles covering 32 keys), K from LDS
    v8f s0 = (v8f){}, s1 = (v8f){};
    {
      const _Float16* kr0 = &sK[lm][0];
      const _Float16* kr1 = &sK[16 + lm][0];
      #pragma unroll
      for (int f = 0; f < 4; f++) {
        v16h bk0 = *(const v16h*)(kr0 + f * 32 + hi * 16);
        s0 = wmma_f16(qf[f].v, bk0, s0);
        v16h bk1 = *(const v16h*)(kr1 + f * 32 + hi * 16);
        s1 = wmma_f16(qf[f].v, bk1, s1);
      }
    }

    // ---- scale + mask + online softmax (per-row over 32 columns)
    const int jg0 = kt * 32 + lm;
    const int jg1 = jg0 + 16;
    #pragma unroll
    for (int r = 0; r < 8; r++) {
      const int ig = qb + hi * 8 + r;
      float x0 = s0[r] * SCALE_;
      float x1 = s1[r] * SCALE_;
      if (ig >= NV_ && jg0 > ig) x0 = NEGBIG;
      if (ig >= NV_ && jg1 > ig) x1 = NEGBIG;

      float mt = fmaxf(x0, x1);
      mt = fmaxf(mt, __shfl_xor(mt, 1, 32));
      mt = fmaxf(mt, __shfl_xor(mt, 2, 32));
      mt = fmaxf(mt, __shfl_xor(mt, 4, 32));
      mt = fmaxf(mt, __shfl_xor(mt, 8, 32));

      float mnew = fmaxf(mrow[r], mt);
      float corr = __expf(mrow[r] - mnew);
      float p0 = __expf(x0 - mnew);
      float p1 = __expf(x1 - mnew);
      float ps = p0 + p1;
      ps += __shfl_xor(ps, 1, 32);
      ps += __shfl_xor(ps, 2, 32);
      ps += __shfl_xor(ps, 4, 32);
      ps += __shfl_xor(ps, 8, 32);
      lrow[r] = lrow[r] * corr + ps;
      mrow[r] = mnew;

      #pragma unroll
      for (int f = 0; f < 8; f++) o[f][r] *= corr;

      const int il = hi * 8 + r;
      sP[wave][il * 32 + lm]      = (_Float16)p0;
      sP[wave][il * 32 + 16 + lm] = (_Float16)p1;
    }
    __syncthreads();

    // ---- P (16x32) as A-fragment from LDS
    F16Frag pf;
    {
      const v8h* pp = (const v8h*)(&sP[wave][lm * 32 + kb]);
      pf.h[0] = pp[0];
      pf.h[1] = pp[1];
    }

    // ---- O += P * V  (8 head-dim slices of 16), V^T from LDS
    #pragma unroll
    for (int sl = 0; sl < 8; sl++) {
      const int ndh = sl * 16 + lm;
      v16h bv = *(const v16h*)(&sVt[ndh][hi * 16]);
      o[sl] = wmma_f16(pf.v, bv, o[sl]);
    }
    __syncthreads();   // all reads of sK/sVt/sP done before next stage overwrites
  }

  // ---- normalize + store f16 attention output (B,T,NH*DH)
  float inv[8];
  #pragma unroll
  for (int r = 0; r < 8; r++) inv[r] = 1.0f / lrow[r];
  #pragma unroll
  for (int sl = 0; sl < 8; sl++) {
    #pragma unroll
    for (int r = 0; r < 8; r++) {
      size_t row = (size_t)(b * T_) + qb + hi * 8 + r;
      attn_h[row * DM_ + h * DH_ + sl * 16 + lm] = (_Float16)(o[sl][r] * inv[r]);
    }
  }
}

// ---------------------------------------------------------------- host launch
extern "C" void kernel_launch(void* const* d_in, const int* in_sizes, int n_in,
                              void* d_out, int out_size, void* d_ws, size_t ws_size,
                              hipStream_t stream) {
  (void)in_sizes; (void)n_in; (void)out_size; (void)ws_size;
  const float* x    = (const float*)d_in[0];
  const float* cosb = (const float*)d_in[1];
  const float* sinb = (const float*)d_in[2];
  const float* wq   = (const float*)d_in[3];
  const float* wk   = (const float*)d_in[4];
  const float* wv   = (const float*)d_in[5];
  const float* wo   = (const float*)d_in[6];

  float* out = (float*)d_out;
  const size_t OUT_K = (size_t)B_ * T_ * DM_;                       // 8388608
  const size_t OUT_V = OUT_K + (size_t)B_ * NKV_ * T_ * DH_;        // +2097152
  float* kcache = out + OUT_K;
  float* vcache = out + OUT_V;

  char* ws = (char*)d_ws;
  size_t off = 0;
  auto alloc = [&](size_t bytes) { char* p = ws + off; off += (bytes + 255) & ~(size_t)255; return p; };

  const size_t M   = (size_t)B_ * T_;          // 4096
  _Float16* x_h   = (_Float16*)alloc(M * DM_ * 2);
  _Float16* wqT   = (_Float16*)alloc((size_t)DM_ * DM_ * 2);
  _Float16* wkT   = (_Float16*)alloc((size_t)DM_ * NKV_ * DH_ * 2);
  _Float16* wvT   = (_Float16*)alloc((size_t)DM_ * NKV_ * DH_ * 2);
  _Float16* woT   = (_Float16*)alloc((size_t)DM_ * DM_ * 2);
  float*    q_f   = (float*)   alloc(M * DM_ * 4);
  float*    k_f   = (float*)   alloc(M * NKV_ * DH_ * 4);
  float*    v_f   = (float*)   alloc(M * NKV_ * DH_ * 4);
  _Float16* q_h   = (_Float16*)alloc(M * DM_ * 2);
  _Float16* k_h   = (_Float16*)alloc(M * NKV_ * DH_ * 2);
  _Float16* v_t   = (_Float16*)alloc(M * NKV_ * DH_ * 2);
  _Float16* a_h   = (_Float16*)alloc(M * DM_ * 2);

  const int TPB = 256;

  // 1) casts
  {
    int n = (int)(M * DM_);
    cast_f32_to_f16<<<dim3(n / TPB), dim3(TPB), 0, stream>>>(x, x_h, n);
  }
  castT_f32_to_f16<<<dim3((DM_ * DM_) / TPB),        dim3(TPB), 0, stream>>>(wq, wqT, DM_, DM_);
  castT_f32_to_f16<<<dim3((DM_ * NKV_ * DH_) / TPB), dim3(TPB), 0, stream>>>(wk, wkT, DM_, NKV_ * DH_);
  castT_f32_to_f16<<<dim3((DM_ * NKV_ * DH_) / TPB), dim3(TPB), 0, stream>>>(wv, wvT, DM_, NKV_ * DH_);
  castT_f32_to_f16<<<dim3((DM_ * DM_) / TPB),        dim3(TPB), 0, stream>>>(wo, woT, DM_, DM_);

  // 2) Q/K/V projections (WMMA, 32x64 per wave)
  gemm_f16_wmma<<<dim3(DM_ / 256, (int)M / 64), dim3(256), 0, stream>>>(x_h, wqT, q_f, (int)M, DM_, DM_);
  gemm_f16_wmma<<<dim3((NKV_ * DH_) / 256, (int)M / 64), dim3(256), 0, stream>>>(x_h, wkT, k_f, (int)M, NKV_ * DH_, DM_);
  gemm_f16_wmma<<<dim3((NKV_ * DH_) / 256, (int)M / 64), dim3(256), 0, stream>>>(x_h, wvT, v_f, (int)M, NKV_ * DH_, DM_);

  // 3) RoPE + packing (+ f32 KV-cache into d_out)
  rope_q_kernel<<<dim3((B_ * T_ * NH_ * 64) / TPB),  dim3(TPB), 0, stream>>>(q_f, cosb, sinb, q_h);
  rope_k_kernel<<<dim3((B_ * T_ * NKV_ * 64) / TPB), dim3(TPB), 0, stream>>>(k_f, cosb, sinb, k_h, kcache);
  v_pack_kernel<<<dim3((B_ * T_ * NKV_ * DH_) / TPB), dim3(TPB), 0, stream>>>(v_f, v_t, vcache);

  // 4) flash attention (WMMA + async LDS staging)
  attn_kernel<<<dim3(T_ / 128, NH_, B_), dim3(256), 0, stream>>>(q_h, k_h, v_t, a_h);

  // 5) output projection (WMMA) -> d_out
  gemm_f16_wmma<<<dim3(DM_ / 256, (int)M / 64), dim3(256), 0, stream>>>(a_h, woT, out, (int)M, DM_, DM_);
}
